// GCN_51170240364741
// MI455X (gfx1250) — matrix operand order repaired and checked by reading, not compiled, not run on hardware
//
#include <hip/hip_runtime.h>
#include <hip/hip_bf16.h>

typedef __attribute__((ext_vector_type(16))) __bf16 v16bf;
typedef __attribute__((ext_vector_type(8)))  __bf16 v8bf;
typedef __attribute__((ext_vector_type(8)))  float  v8f;

#define CH 128          // feature channels (K of every GEMM)
#define AS_STRIDE 136   // LDS row stride in bf16 elems (pad 128 -> 136 kills bank conflicts)

// ---------------------------------------------------------------- small utils

__global__ void deg_init_kernel(float* dinv, int n) {
    int i = blockIdx.x * blockDim.x + threadIdx.x;
    if (i < n) dinv[i] = 1.0f;                 // self-loop counts as 1
}

__global__ void deg_edge_kernel(const int* __restrict__ row, float* dinv, int E) {
    int e = blockIdx.x * blockDim.x + threadIdx.x;
    if (e < E) unsafeAtomicAdd(&dinv[row[e]], 1.0f);
}

__global__ void deg_fin_kernel(float* dinv, int n) {
    int i = blockIdx.x * blockDim.x + threadIdx.x;
    if (i < n) dinv[i] = rsqrtf(fmaxf(dinv[i], 1.0f));
}

__global__ void cvt_bf16_kernel(const float* __restrict__ x, __bf16* __restrict__ o, long long n) {
    long long i = (long long)blockIdx.x * blockDim.x + threadIdx.x;
    if (i < n) o[i] = (__bf16)x[i];
}

// WT[n][k] = W[k][n] (bf16, zero-padded columns)
__global__ void pack_wt_kernel(const float* __restrict__ W, __bf16* __restrict__ WT,
                               int ncols, int npad) {
    int idx = blockIdx.x * blockDim.x + threadIdx.x;
    if (idx >= npad * CH) return;
    int n = idx / CH, k = idx % CH;
    float v = (n < ncols) ? W[(size_t)k * ncols + n] : 0.0f;
    WT[(size_t)n * CH + k] = (__bf16)v;
}

// ---------------------------------------------------------------- WMMA core
// Two 16x16 f32 tiles (rows m and m+16) sharing the same B fragment.
// 8x v_wmma_f32_16x16x32_bf16 per call; K = 128.
__device__ __forceinline__ void wmma_k128_x2(const __bf16* aRow0, const __bf16* aRow1,
                                             const __bf16* __restrict__ bRow, int half,
                                             v8f& c0, v8f& c1) {
    #pragma unroll
    for (int s = 0; s < 4; ++s) {
        // B 16-bit 32x16 layout: lane<16 holds K = 0..7,16..23 ; lane>=16 holds K = 8..15,24..31 (col n)
        v8bf blo = *(const v8bf*)(bRow + 32 * s + 8 * half);
        v8bf bhi = *(const v8bf*)(bRow + 32 * s + 16 + 8 * half);
        // A 16-bit 16x32 layout: lane<16 holds K = 0..7,16..23 ; lane>=16 holds K = 8..15,24..31
        v8bf a0lo = *(const v8bf*)(aRow0 + 32 * s + 8 * half);
        v8bf a0hi = *(const v8bf*)(aRow0 + 32 * s + 16 + 8 * half);
        v8bf a1lo = *(const v8bf*)(aRow1 + 32 * s + 8 * half);
        v8bf a1hi = *(const v8bf*)(aRow1 + 32 * s + 16 + 8 * half);
        v16bf a0, a1, b;
        #pragma unroll
        for (int i = 0; i < 8; ++i) {
            b[i]  = blo[i];  b[i + 8]  = bhi[i];
            a0[i] = a0lo[i]; a0[i + 8] = a0hi[i];
            a1[i] = a1lo[i]; a1[i + 8] = a1hi[i];
        }
        c0 = __builtin_amdgcn_wmma_f32_16x16x32_bf16(false, a0, false, b,
                                                     (short)0, c0, false, false);
        c1 = __builtin_amdgcn_wmma_f32_16x16x32_bf16(false, a1, false, b,
                                                     (short)0, c1, false, false);
    }
}

// stage a 32-row x 128-col bf16 strip into LDS, coalesced, rows clamped to nrows
__device__ __forceinline__ void load_strip32(const __bf16* __restrict__ A,
                                             __bf16 (*As)[AS_STRIDE],
                                             int row0, int nrows, int t) {
    int r = t >> 4, chunk = t & 15;
    #pragma unroll
    for (int rr = 0; rr < 2; ++rr) {
        int row = row0 + r + 16 * rr;
        if (row >= nrows) row = nrows - 1;
        *(v8bf*)&As[r + 16 * rr][chunk * 8] =
            *(const v8bf*)(A + (size_t)row * CH + chunk * 8);
    }
}

// ---------------------------------------------------------------- node GEMM (Ncols = 128)
// grid.x = 32-row tiles ; block = 256 (8 waves, one 16-col tile each -> full 128 cols)
__global__ __launch_bounds__(256) void gemm_nodes_kernel(const __bf16* __restrict__ A,
                                                         const __bf16* __restrict__ BT,
                                                         float* __restrict__ out, int nrows) {
    __shared__ __bf16 As[32][AS_STRIDE];
    int t = threadIdx.x;
    int row0 = blockIdx.x * 32;
    load_strip32(A, As, row0, nrows, t);
    __syncthreads();

    int wave = t >> 5, lane = t & 31;
    int m = lane & 15, half = lane >> 4;
    int n0 = wave * 16;
    v8f c0 = {}, c1 = {};
    wmma_k128_x2(&As[m][0], &As[m + 16][0], BT + (size_t)(n0 + m) * CH, half, c0, c1);

    int col = n0 + m;
    int rb0 = row0 + 8 * half;          // C layout: elem i -> row rb + i
    int rb1 = row0 + 16 + 8 * half;
    if (row0 + 32 <= nrows) {           // fast path: no per-row guards
        #pragma unroll
        for (int i = 0; i < 8; ++i) out[(size_t)(rb0 + i) * CH + col] = c0[i];
        #pragma unroll
        for (int i = 0; i < 8; ++i) out[(size_t)(rb1 + i) * CH + col] = c1[i];
    } else {
        #pragma unroll
        for (int i = 0; i < 8; ++i)
            if (rb0 + i < nrows) out[(size_t)(rb0 + i) * CH + col] = c0[i];
        #pragma unroll
        for (int i = 0; i < 8; ++i)
            if (rb1 + i < nrows) out[(size_t)(rb1 + i) * CH + col] = c1[i];
    }
}

// ---------------------------------------------------------------- fused heads GEMM
__global__ __launch_bounds__(256) void gemm_heads_kernel(
    const __bf16* __restrict__ A, const __bf16* __restrict__ BT, float* __restrict__ out,
    const float* __restrict__ bt, const float* __restrict__ bs,
    const float* __restrict__ bfv, const float* __restrict__ ba,
    int nrows, int P, int pS1, int pS2, int pS3,
    int w0, int w1, int w2, int w3,
    long long base1, long long base2, long long base3) {
    __shared__ __bf16 As[32][AS_STRIDE];
    int t = threadIdx.x;
    int row0 = blockIdx.x * 32;
    load_strip32(A, As, row0, nrows, t);
    __syncthreads();

    int wave = t >> 5, lane = t & 31;
    int ptile = blockIdx.y * 8 + wave;
    if (ptile * 16 >= P) return;
    int m = lane & 15, half = lane >> 4;
    int n0 = ptile * 16;
    v8f c0 = {}, c1 = {};
    wmma_k128_x2(&As[m][0], &As[m + 16][0], BT + (size_t)(n0 + m) * CH, half, c0, c1);

    int p = n0 + m;                       // padded column -> (head, in-head col)
    long long base; int w, cc; const float* bias;
    if (p < pS1)      { base = 0;     w = w0; cc = p;       bias = bt;  }
    else if (p < pS2) { base = base1; w = w1; cc = p - pS1; bias = bs;  }
    else if (p < pS3) { base = base2; w = w2; cc = p - pS2; bias = bfv; }
    else              { base = base3; w = w3; cc = p - pS3; bias = ba;  }
    if (cc >= w) return;
    float bv = bias[cc];
    long long o = base + cc;
    int rb0 = row0 + 8 * half;
    int rb1 = row0 + 16 + 8 * half;
    if (row0 + 32 <= nrows) {             // fast path: no per-row guards
        #pragma unroll
        for (int i = 0; i < 8; ++i) out[o + (long long)(rb0 + i) * w] = c0[i] + bv;
        #pragma unroll
        for (int i = 0; i < 8; ++i) out[o + (long long)(rb1 + i) * w] = c1[i] + bv;
    } else {
        #pragma unroll
        for (int i = 0; i < 8; ++i)
            if (rb0 + i < nrows) out[o + (long long)(rb0 + i) * w] = c0[i] + bv;
        #pragma unroll
        for (int i = 0; i < 8; ++i)
            if (rb1 + i < nrows) out[o + (long long)(rb1 + i) * w] = c1[i] + bv;
    }
}

// ---------------------------------------------------------------- edge scatter
// one wave per edge; lane handles 4 channels (float4 gather, 4 f32 atomics -> L2-resident)
__global__ __launch_bounds__(256) void scatter_kernel(const int* __restrict__ ei,
                                                      const float* __restrict__ h,
                                                      const float* __restrict__ dinv,
                                                      float* __restrict__ agg, int E) {
    long long tid = (long long)blockIdx.x * blockDim.x + threadIdx.x;
    long long e = tid >> 5;
    if (e >= E) return;
    int lane = (int)(tid & 31);
    int rw = ei[e];
    int cl = ei[(long long)E + e];
    float norm = dinv[rw] * dinv[cl];
    const float4 mv = ((const float4*)(h + (size_t)cl * CH))[lane];
    float* dst = agg + (size_t)rw * CH + lane * 4;
    unsafeAtomicAdd(dst + 0, mv.x * norm);
    unsafeAtomicAdd(dst + 1, mv.y * norm);
    unsafeAtomicAdd(dst + 2, mv.z * norm);
    unsafeAtomicAdd(dst + 3, mv.w * norm);
}

// out = agg + prod * dinv^2 (self loop) + bias, optional relu; emit bf16 for next GEMM
__global__ void finalize_kernel(const float* __restrict__ agg, const float* __restrict__ prod,
                                const float* __restrict__ dinv, const float* __restrict__ bias,
                                __bf16* __restrict__ obf, int n, int relu) {
    long long idx = (long long)blockIdx.x * blockDim.x + threadIdx.x;
    if (idx >= (long long)n * CH) return;
    int i = (int)(idx / CH), c = (int)(idx % CH);
    float dv = dinv[i];
    float v = agg[idx] + prod[idx] * dv * dv + bias[c];
    if (relu) v = fmaxf(v, 0.0f);
    obf[idx] = (__bf16)v;
}

// ---------------------------------------------------------------- launcher
extern "C" void kernel_launch(void* const* d_in, const int* in_sizes, int n_in,
                              void* d_out, int out_size, void* d_ws, size_t ws_size,
                              hipStream_t stream) {
    const float* x   = (const float*)d_in[0];
    const int*   ei  = (const int*)d_in[1];
    const float* W1  = (const float*)d_in[2];
    const float* b1  = (const float*)d_in[3];
    const float* W2  = (const float*)d_in[4];
    const float* b2  = (const float*)d_in[5];
    const float* Wt  = (const float*)d_in[6];  const float* bt  = (const float*)d_in[7];
    const float* Ws  = (const float*)d_in[8];  const float* bs  = (const float*)d_in[9];
    const float* Wf  = (const float*)d_in[10]; const float* bfv = (const float*)d_in[11];
    const float* Wa  = (const float*)d_in[12]; const float* ba  = (const float*)d_in[13];

    int N = in_sizes[0] / CH;
    int E = in_sizes[1] / 2;
    int w0 = in_sizes[6] / CH, w1 = in_sizes[8] / CH, w2 = in_sizes[10] / CH, w3 = in_sizes[12] / CH;
    int p0 = (w0 + 15) & ~15, p1 = (w1 + 15) & ~15, p2 = (w2 + 15) & ~15, p3 = (w3 + 15) & ~15;
    int P = p0 + p1 + p2 + p3;
    int pS1 = p0, pS2 = p0 + p1, pS3 = p0 + p1 + p2;
    long long base1 = (long long)N * w0;
    long long base2 = base1 + (long long)N * w1;
    long long base3 = base2 + (long long)N * w2;

    auto al = [](size_t v) { return (v + 255) & ~(size_t)255; };
    char* ws = (char*)d_ws;
    size_t off = 0;
    float*  dinv   = (float*)(ws + off);  off += al((size_t)N * 4);
    __bf16* featbf = (__bf16*)(ws + off); off += al((size_t)N * CH * 2);
    float*  bufP   = (float*)(ws + off);  off += al((size_t)N * CH * 4);
    float*  bufG   = (float*)(ws + off);  off += al((size_t)N * CH * 4);
    __bf16* W1T    = (__bf16*)(ws + off); off += al((size_t)CH * CH * 2);
    __bf16* W2T    = (__bf16*)(ws + off); off += al((size_t)CH * CH * 2);
    __bf16* WhT    = (__bf16*)(ws + off); off += al((size_t)P * CH * 2);
    (void)ws_size; (void)n_in; (void)out_size;

    const int B = 256;
    long long nElem = (long long)N * CH;

    // symmetric normalization: deg over destinations (+self loop), dinv = rsqrt
    deg_init_kernel<<<(N + B - 1) / B, B, 0, stream>>>(dinv, N);
    deg_edge_kernel<<<(E + B - 1) / B, B, 0, stream>>>(ei, dinv, E);
    deg_fin_kernel<<<(N + B - 1) / B, B, 0, stream>>>(dinv, N);

    // bf16 conversions / weight packing (transposed, zero-padded heads)
    cvt_bf16_kernel<<<(int)((nElem + B - 1) / B), B, 0, stream>>>(x, featbf, nElem);
    pack_wt_kernel<<<(CH * CH + B - 1) / B, B, 0, stream>>>(W1, W1T, CH, CH);
    pack_wt_kernel<<<(CH * CH + B - 1) / B, B, 0, stream>>>(W2, W2T, CH, CH);
    pack_wt_kernel<<<(p0 * CH + B - 1) / B, B, 0, stream>>>(Wt, WhT, w0, p0);
    pack_wt_kernel<<<(p1 * CH + B - 1) / B, B, 0, stream>>>(Ws, WhT + (size_t)pS1 * CH, w1, p1);
    pack_wt_kernel<<<(p2 * CH + B - 1) / B, B, 0, stream>>>(Wf, WhT + (size_t)pS2 * CH, w2, p2);
    pack_wt_kernel<<<(p3 * CH + B - 1) / B, B, 0, stream>>>(Wa, WhT + (size_t)pS3 * CH, w3, p3);

    int rowTiles = (N + 31) / 32;
    long long sThreads = (long long)E * 32;

    // ---- layer 1: h1 = relu(Aggr(x @ W1) + b1)
    gemm_nodes_kernel<<<rowTiles, 256, 0, stream>>>(featbf, W1T, bufP, N);
    hipMemsetAsync(bufG, 0, (size_t)N * CH * 4, stream);
    scatter_kernel<<<(int)((sThreads + B - 1) / B), B, 0, stream>>>(ei, bufP, dinv, bufG, E);
    finalize_kernel<<<(int)((nElem + B - 1) / B), B, 0, stream>>>(bufG, bufP, dinv, b1, featbf, N, 1);

    // ---- layer 2: emb = Aggr(h1 @ W2) + b2
    gemm_nodes_kernel<<<rowTiles, 256, 0, stream>>>(featbf, W2T, bufP, N);
    hipMemsetAsync(bufG, 0, (size_t)N * CH * 4, stream);
    scatter_kernel<<<(int)((sThreads + B - 1) / B), B, 0, stream>>>(ei, bufP, dinv, bufG, E);
    finalize_kernel<<<(int)((nElem + B - 1) / B), B, 0, stream>>>(bufG, bufP, dinv, b2, featbf, N, 0);

    // ---- fused 4-head classifier GEMM straight into d_out
    dim3 hg(rowTiles, (P / 16 + 7) / 8);
    gemm_heads_kernel<<<hg, 256, 0, stream>>>(featbf, WhT, (float*)d_out,
                                              bt, bs, bfv, ba, N, P, pS1, pS2, pS3,
                                              w0, w1, w2, w3, base1, base2, base3);
}
